// QuantLlamaMLP_549755813920
// MI455X (gfx1250) — matrix operand order repaired and checked by reading, not compiled, not run on hardware
//
#include <hip/hip_runtime.h>

// ---------------------------------------------------------------------------
// AWQ int4 Llama MLP for MI455X (gfx1250, wave32, WMMA + async-to-LDS).
//   h   = silu(x @ Wg) * (x @ Wu)     (Wg, Wu int4-dequant, K=4096 -> I=11008)
//   out = h @ Wd                      (Wd int4-dequant,   I=11008 -> O=4096)
// Memory-bound: ~290MB HBM traffic vs ~1.1 TFLOP of f16 WMMA math.
// x converted to f16 once (33.5MB, L2-resident); h kept as f16 in ws.
// Activations stream into LDS via GLOBAL_LOAD_ASYNC_TO_LDS (ASYNCcnt),
// overlapping with packed-f16 (v_pk_fma_f16) int4 dequant of weight tiles.
// ---------------------------------------------------------------------------

typedef __attribute__((ext_vector_type(16))) _Float16 v16h;
typedef __attribute__((ext_vector_type(8)))  _Float16 v8h;
typedef __attribute__((ext_vector_type(4)))  _Float16 v4h;
typedef __attribute__((ext_vector_type(2)))  _Float16 h2;
typedef __attribute__((ext_vector_type(8)))  float    v8f;

// 64-bit async payload type: matches the builtin's declared pointee type.
typedef int v2i_pay __attribute__((vector_size(8)));

#define AWQ_GROUP 128
#define DK 4096      // in_features
#define DI 11008     // intermediate_size
#define DO 4096      // out_features

#if __has_builtin(__builtin_amdgcn_global_load_async_to_lds_b64)
#define HAVE_ASYNC_LDS 1
#else
#define HAVE_ASYNC_LDS 0
#endif

#define AS_GLOBAL __attribute__((address_space(1)))
#define AS_LDS    __attribute__((address_space(3)))

// A-tile element move: global f16x4 (8B) -> LDS via GLOBAL_LOAD_ASYNC_TO_LDS_B64
// (tracked by ASYNCcnt, no VGPR staging).
__device__ __forceinline__ void load8_to_lds(const _Float16* gsrc, _Float16* ldst) {
#if HAVE_ASYNC_LDS
  __builtin_amdgcn_global_load_async_to_lds_b64(
      (AS_GLOBAL v2i_pay*)gsrc, (AS_LDS v2i_pay*)ldst, 0, 0);
#else
  *(v4h*)ldst = *(const v4h*)gsrc;
#endif
}

__device__ __forceinline__ void async_wait_all() {
#if HAVE_ASYNC_LDS
#if __has_builtin(__builtin_amdgcn_s_wait_asynccnt)
  __builtin_amdgcn_s_wait_asynccnt(0);
#else
  asm volatile("s_wait_asynccnt 0" ::: "memory");
#endif
#endif
}

// ---------------- fp32 -> f16 convert (x stays hot in L2 as f16) -----------
__global__ void cvt_kernel(const float* __restrict__ x, _Float16* __restrict__ y, int n) {
  int i = (blockIdx.x * blockDim.x + threadIdx.x) * 4;
  if (i < n) {
    float4 v = *(const float4*)(x + i);
    v4h h;
    h.x = (_Float16)v.x; h.y = (_Float16)v.y;
    h.z = (_Float16)v.z; h.w = (_Float16)v.w;
    *(v4h*)(y + i) = h;
  }
}

// Pack two 8-half LDS reads (b128 each) into one WMMA 16-half operand.
__device__ __forceinline__ v16h pack16(const _Float16* lo, const _Float16* hi) {
  v8h a = *(const v8h*)lo;
  v8h b = *(const v8h*)hi;
  v16h r;
#pragma unroll
  for (int j = 0; j < 8; ++j) { r[j] = a[j]; r[8 + j] = b[j]; }
  return r;
}

// Packed dequant of one nibble-column pair (two k's, same column => same scale):
// one v_pk_fma_f16 + one ds_store_b32.
__device__ __forceinline__ void dq_pair(int w0, int w1, int i,
                                        const _Float16* ss, const _Float16* bb,
                                        _Float16* Bs, int ncol_base, int k2) {
  int n = ncol_base + i;
  h2 q2;
  q2.x = (_Float16)(unsigned short)((w0 >> (4 * i)) & 0xF);
  q2.y = (_Float16)(unsigned short)((w1 >> (4 * i)) & 0xF);
  h2 s2; s2.x = ss[n]; s2.y = s2.x;
  h2 b2; b2.x = bb[n]; b2.y = b2.x;
  *(h2*)&Bs[n * 32 + k2] = q2 * s2 + b2;      // [n][k] col-major, k-pair adjacent
}

// ---------------- fused gate/up GEMM + SiLU*mul epilogue --------------------
// Block: 256 thr = 8 waves (2 M-waves x 4 N-waves). Tile BM=32, BN=64, BK=32.
__launch_bounds__(256, 2)
__global__ void gateup_kernel(const _Float16* __restrict__ xh,
                              const int*   __restrict__ gqw,
                              const float* __restrict__ gsc,
                              const int*   __restrict__ gqz,
                              const int*   __restrict__ uqw,
                              const float* __restrict__ usc,
                              const int*   __restrict__ uqz,
                              _Float16* __restrict__ hout) {
  __shared__ __attribute__((aligned(16))) _Float16 As[32 * 32];   // [m][k]
  __shared__ __attribute__((aligned(16))) _Float16 Bg[64 * 32];   // [n][k] col-major
  __shared__ __attribute__((aligned(16))) _Float16 Bu[64 * 32];   // [n][k] col-major
  __shared__ _Float16 sG[64], bG[64], sU[64], bU[64];             // f16 scale / -z*scale

  const int WPR = DI / 8;               // packed words per K-row
  const int n0 = blockIdx.x * 64;
  const int m0 = blockIdx.y * 32;

  const int t    = threadIdx.x;
  const int lane = t & 31;
  const int w    = t >> 5;
  const int wm   = w >> 2;              // 0..1  (M sub-tile)
  const int wn   = w & 3;               // 0..3  (N sub-tile)

  const int arow = t >> 3;              // 0..31 : A-tile loader row
  const int acol = (t & 7) * 4;         //        4 halfs each

  // Dequant split: threads 0-127 -> gate, 128-255 -> up (wave-uniform).
  const int dq_mat = t >> 7;
  const int dq_id  = t & 127;
  const int kp     = dq_id >> 3;        // 0..15 k-pair index
  const int dwc    = dq_id & 7;         // packed word column

  v8f accG = {};
  v8f accU = {};

  for (int ks = 0; ks < DK; ks += 32) {
    __syncthreads();                    // WAR: previous compute done with LDS

    if ((ks & (AWQ_GROUP - 1)) == 0) {  // new quant group: refresh scale/bias
      if (t < 128) {
        const int   which = t >> 6;     // 0 = gate, 1 = up
        const int   n     = t & 63;
        const float* sc = which ? usc : gsc;
        const int*   qz = which ? uqz : gqz;
        const int    g  = ks / AWQ_GROUP;
        float s  = sc[(size_t)g * DI + n0 + n];
        int   zw = qz[(size_t)g * WPR + ((n0 + n) >> 3)];
        float z  = (float)((zw >> (4 * (n & 7))) & 0xF);
        if (which) { sU[n] = (_Float16)s; bU[n] = (_Float16)(-z * s); }
        else       { sG[n] = (_Float16)s; bG[n] = (_Float16)(-z * s); }
      }
      __syncthreads();
    }

    // A tile 32x32 f16: async global->LDS (overlaps with weight dequant below)
    load8_to_lds(xh + (size_t)(m0 + arow) * DK + ks + acol,
                 &As[arow * 32 + acol]);

    // B tiles: packed-f16 dequant, one matrix per thread-half, one k-pair each.
    {
      const int*      qw = dq_mat ? uqw : gqw;
      _Float16*       Bs = dq_mat ? Bu : Bg;
      const _Float16* ss = dq_mat ? sU : sG;
      const _Float16* bb = dq_mat ? bU : bG;
      size_t rbase = (size_t)(ks + 2 * kp) * WPR + (n0 >> 3) + dwc;
      int w0 = qw[rbase];
      int w1 = qw[rbase + WPR];
#pragma unroll
      for (int i = 0; i < 8; ++i)
        dq_pair(w0, w1, i, ss, bb, Bs, dwc * 8, 2 * kp);
      if (ks + 32 < DK)                 // gfx1250 global_prefetch_b8
        __builtin_prefetch(qw + rbase + 32ull * WPR, 0, 1);
    }
    async_wait_all();                   // s_wait_asynccnt 0 (A tile landed)
    __syncthreads();

    // ---- WMMA fragments per ISA 16-bit layouts ----
    const int lm = lane & 15;
    const int kh = lane >> 4;
    // A 16x32: lane row = lane%16; V0-3 K=kh*8+0..7, V4-7 K=16+kh*8+0..7
    const _Float16* ap = &As[(wm * 16 + lm) * 32 + kh * 8];
    v16h a = pack16(ap, ap + 16);
    // B 32x16: lane col = lane%16; V0-3 K=kh*8+0..7, V4-7 K=16+kh*8+0..7 (col-major LDS)
    const _Float16* bgp = &Bg[(wn * 16 + lm) * 32 + kh * 8];
    v16h bg = pack16(bgp, bgp + 16);
    const _Float16* bup = &Bu[(wn * 16 + lm) * 32 + kh * 8];
    v16h bu = pack16(bup, bup + 16);

    accG = __builtin_amdgcn_wmma_f32_16x16x32_f16(false, a, false, bg,
                                                  (short)0, accG, false, false);
    accU = __builtin_amdgcn_wmma_f32_16x16x32_f16(false, a, false, bu,
                                                  (short)0, accU, false, false);
  }

  // Epilogue: h = silu(g) * u, written f16. C/D layout: VGPR r -> M=(lane/16)*8+r, N=lane%16.
  {
    const int    lm   = lane & 15;
    const int    mtop = (lane >> 4) * 8;
    const size_t col  = (size_t)n0 + wn * 16 + lm;
#pragma unroll
    for (int r = 0; r < 8; ++r) {
      float g  = accG[r];
      float u  = accU[r];
      float sg = g / (1.0f + __expf(-g));
      size_t row = (size_t)m0 + wm * 16 + mtop + r;
      hout[row * DI + col] = (_Float16)(sg * u);
    }
  }
}

// ---------------- down GEMM: out = h @ Wd (fp32 out) ------------------------
__launch_bounds__(256, 2)
__global__ void down_kernel(const _Float16* __restrict__ hh,
                            const int*   __restrict__ dqw,
                            const float* __restrict__ dsc,
                            const int*   __restrict__ dqz,
                            float* __restrict__ out) {
  __shared__ __attribute__((aligned(16))) _Float16 As[32 * 32];
  __shared__ __attribute__((aligned(16))) _Float16 Bd[64 * 32];
  __shared__ _Float16 sD[64], bD[64];

  const int WPR = DO / 8;
  const int n0 = blockIdx.x * 64;
  const int m0 = blockIdx.y * 32;

  const int t    = threadIdx.x;
  const int lane = t & 31;
  const int w    = t >> 5;
  const int wm   = w >> 2;
  const int wn   = w & 3;

  const int arow = t >> 3;
  const int acol = (t & 7) * 4;

  // Dequant split: 16 k-pairs x 8 word-cols x 2 nibble-halves = 256 tasks.
  const int kp  = t >> 4;               // 0..15
  const int dwc = (t >> 1) & 7;         // 0..7
  const int ih  = t & 1;                // nibble half (0: i=0..3, 1: i=4..7)

  v8f acc = {};

  for (int ks = 0; ks < DI; ks += 32) {
    __syncthreads();

    if ((ks & (AWQ_GROUP - 1)) == 0) {
      if (t < 64) {
        const int n = t;
        const int g = ks / AWQ_GROUP;
        float s  = dsc[(size_t)g * DO + n0 + n];
        int   zw = dqz[(size_t)g * WPR + ((n0 + n) >> 3)];
        float z  = (float)((zw >> (4 * (n & 7))) & 0xF);
        sD[n] = (_Float16)s; bD[n] = (_Float16)(-z * s);
      }
      __syncthreads();
    }

    load8_to_lds(hh + (size_t)(m0 + arow) * DI + ks + acol,
                 &As[arow * 32 + acol]);

    {
      size_t rbase = (size_t)(ks + 2 * kp) * WPR + (n0 >> 3) + dwc;
      int w0 = dqw[rbase];
      int w1 = dqw[rbase + WPR];
#pragma unroll
      for (int j = 0; j < 4; ++j)
        dq_pair(w0, w1, ih * 4 + j, sD, bD, Bd, dwc * 8, 2 * kp);
      if (ks + 32 < DI)
        __builtin_prefetch(dqw + rbase + 32ull * WPR, 0, 1);
    }
    async_wait_all();
    __syncthreads();

    const int lm = lane & 15;
    const int kh = lane >> 4;
    const _Float16* ap = &As[(wm * 16 + lm) * 32 + kh * 8];
    v16h a = pack16(ap, ap + 16);
    const _Float16* bp = &Bd[(wn * 16 + lm) * 32 + kh * 8];
    v16h b = pack16(bp, bp + 16);

    acc = __builtin_amdgcn_wmma_f32_16x16x32_f16(false, a, false, b,
                                                 (short)0, acc, false, false);
  }

  {
    const int    lm   = lane & 15;
    const int    mtop = (lane >> 4) * 8;
    const size_t col  = (size_t)n0 + wn * 16 + lm;
#pragma unroll
    for (int r = 0; r < 8; ++r) {
      size_t row = (size_t)m0 + wm * 16 + mtop + r;
      out[row * DO + col] = acc[r];
    }
  }
}

// ---------------------------------------------------------------------------
extern "C" void kernel_launch(void* const* d_in, const int* in_sizes, int n_in,
                              void* d_out, int out_size, void* d_ws, size_t ws_size,
                              hipStream_t stream) {
  const float* x   = (const float*)d_in[0];
  const int*   gqw = (const int*)d_in[1];
  const float* gsc = (const float*)d_in[2];
  const int*   gqz = (const int*)d_in[3];
  const int*   uqw = (const int*)d_in[4];
  const float* usc = (const float*)d_in[5];
  const int*   uqz = (const int*)d_in[6];
  const int*   dqw = (const int*)d_in[7];
  const float* dsc = (const float*)d_in[8];
  const int*   dqz = (const int*)d_in[9];
  float* out = (float*)d_out;

  const int T = in_sizes[0] / DK;       // 4096 tokens

  // workspace: [x as f16 : T*DK] [h as f16 : T*DI]   (~124 MB total)
  char* wsb = (char*)d_ws;
  _Float16* xh = (_Float16*)wsb;
  _Float16* h  = (_Float16*)(wsb + (size_t)T * DK * sizeof(_Float16));

  {
    int n = T * DK;
    cvt_kernel<<<n / (256 * 4), 256, 0, stream>>>(x, xh, n);
  }
  gateup_kernel<<<dim3(DI / 64, T / 32), 256, 0, stream>>>(
      xh, gqw, gsc, gqz, uqw, usc, uqz, h);
  down_kernel<<<dim3(DO / 64, T / 32), 256, 0, stream>>>(
      h, dqw, dsc, dqz, out);
}
